// Backbone_82403242541513
// MI455X (gfx1250) — matrix operand order repaired and checked by reading, not compiled, not run on hardware
//
// Point Transformer backbone for MI455X (gfx1250), wave32 + WMMA f16->f32.
// All heavy linears (fc1/wq/wk/wv/d2/g1/g2/fc2, TD convs with folded BN) run
// through one LDS-staged v_wmma_f32_16x16x32_f16 GEMM kernel with fused
// scale/bias/relu/residual epilogue and dual f32/f16 output. The GEMM uses
// clamped (branch-free) global->LDS tile loads and double-buffered LDS with
// register staging so global loads overlap the WMMA stream. Intermediates are
// kept in f16 to halve traffic against the 23.3 TB/s HBM roof. KNN / softmax /
// gathers / maxpool are VALU kernels (they are a small fraction of the work).
#include <hip/hip_runtime.h>

typedef __attribute__((ext_vector_type(16))) _Float16 v16h;
typedef __attribute__((ext_vector_type(8)))  float    v8f;

#define TM 128
#define TN 128
#define TKS 32
#define LDP 40   // LDS row stride in halfs (32 + 8 pad) to avoid bank conflicts

// ---------------------------------------------------------------- GEMM (WMMA)
// C[M,N] = act( (A[M,K] @ W) * scale + bias ) + resid ; W given transposed as
// WT[N,Kpad] f16. A row stride == K (padded, multiple of 32).
// Out-of-range tile rows/cols are CLAMPED (not zeroed) on load: a clamped A row
// only pollutes C rows >= M and a clamped B row only C cols >= N, both of which
// the epilogue discards. This keeps the load path branch-free (no exec masking).
__global__ __launch_bounds__(256)
void k_gemm(const _Float16* __restrict__ A, const _Float16* __restrict__ WT,
            int M, int N, int K,
            const float* __restrict__ scale, const float* __restrict__ bias,
            const float* __restrict__ resid, int relu,
            float* __restrict__ outF, _Float16* __restrict__ outH)
{
    __shared__ __align__(16) _Float16 sA[2][TM * LDP];
    __shared__ __align__(16) _Float16 sB[2][TN * LDP];
    const int t    = threadIdx.x;
    const int lane = t & 31;
    const int wave = t >> 5;
    const int m0 = blockIdx.x * TM;
    const int n0 = blockIdx.y * TN;
    const int wm = (wave & 3) * 32;   // wave computes 32 rows x 64 cols
    const int wn = (wave >> 2) * 64;
    const int l15 = lane & 15;
    const int hi  = lane >> 4;        // 0 or 1: which K-half this lane holds

    // Per-thread tile-load coordinates: 512 chunks of 8 halfs cover 128x32.
    const int c0 = t, c1 = t + 256;
    const int r0 = c0 >> 2, o0 = (c0 & 3) * 8;
    const int r1 = c1 >> 2, o1 = (c1 & 3) * 8;
    int ga0 = m0 + r0; if (ga0 >= M) ga0 = M - 1;
    int ga1 = m0 + r1; if (ga1 >= M) ga1 = M - 1;
    int gb0 = n0 + r0; if (gb0 >= N) gb0 = N - 1;
    int gb1 = n0 + r1; if (gb1 >= N) gb1 = N - 1;
    const _Float16* pa0 = A  + (size_t)ga0 * K + o0;
    const _Float16* pa1 = A  + (size_t)ga1 * K + o1;
    const _Float16* pb0 = WT + (size_t)gb0 * K + o0;
    const _Float16* pb1 = WT + (size_t)gb1 * K + o1;
    const int dA0 = r0 * LDP + o0, dA1 = r1 * LDP + o1;

    // Prologue: stage K-tile 0 into LDS buffer 0.
    uint4 ra0 = *(const uint4*)(pa0);
    uint4 ra1 = *(const uint4*)(pa1);
    uint4 rb0 = *(const uint4*)(pb0);
    uint4 rb1 = *(const uint4*)(pb1);
    *(uint4*)(sA[0] + dA0) = ra0;
    *(uint4*)(sA[0] + dA1) = ra1;
    *(uint4*)(sB[0] + dA0) = rb0;
    *(uint4*)(sB[0] + dA1) = rb1;
    __syncthreads();

    v8f zero = {};
    v8f acc[2][4];
    for (int a = 0; a < 2; ++a) for (int b = 0; b < 4; ++b) acc[a][b] = zero;

    int cur = 0;
    for (int k0 = 0; k0 < K; k0 += TKS) {
        const bool hasNext = (k0 + TKS) < K;
        if (hasNext) {                       // issue next tile's global loads early
            ra0 = *(const uint4*)(pa0 + k0 + TKS);
            ra1 = *(const uint4*)(pa1 + k0 + TKS);
            rb0 = *(const uint4*)(pb0 + k0 + TKS);
            rb1 = *(const uint4*)(pb1 + k0 + TKS);
        }
        const _Float16* sAc = sA[cur];
        const _Float16* sBc = sB[cur];

        // Build fragments per ISA layouts.
        // A 16x32 f16: lane l<16 row=l holds K 0..7 & 16..23; lane l+16 same row K 8..15 & 24..31.
        v16h aF[2], bF[4];
        for (int a = 0; a < 2; ++a) {
            const _Float16* base = sAc + (wm + a * 16 + l15) * LDP + hi * 8;
            union { v16h v; uint4 q[2]; } u;
            u.q[0] = *(const uint4*)(base);
            u.q[1] = *(const uint4*)(base + 16);
            aF[a] = u.v;
        }
        // B 32x16 f16: lane holds column N=l15, K = hi*8..hi*8+7 & 16+hi*8..16+hi*8+7.
        for (int b = 0; b < 4; ++b) {
            const _Float16* base = sBc + (wn + b * 16 + l15) * LDP + hi * 8;
            union { v16h v; uint4 q[2]; } u;
            u.q[0] = *(const uint4*)(base);
            u.q[1] = *(const uint4*)(base + 16);
            bF[b] = u.v;
        }
        for (int a = 0; a < 2; ++a)
            for (int b = 0; b < 4; ++b)
                acc[a][b] = __builtin_amdgcn_wmma_f32_16x16x32_f16(
                    false, aF[a], false, bF[b], (short)0, acc[a][b], false, false);

        if (hasNext) {                       // store staged tile into other buffer
            _Float16* dAn = sA[cur ^ 1];
            _Float16* dBn = sB[cur ^ 1];
            *(uint4*)(dAn + dA0) = ra0;
            *(uint4*)(dAn + dA1) = ra1;
            *(uint4*)(dBn + dA0) = rb0;
            *(uint4*)(dBn + dA1) = rb1;
        }
        __syncthreads();
        cur ^= 1;
    }

    // Epilogue. C layout: lane => col=l15, rows (hi*8 .. hi*8+7) of its 16-row tile.
    for (int b = 0; b < 4; ++b) {
        int col = n0 + wn + b * 16 + l15;
        if (col >= N) continue;
        float sc = scale ? scale[col] : 1.f;
        float bi = bias  ? bias[col]  : 0.f;
        for (int a = 0; a < 2; ++a) {
            int rbase = m0 + wm + a * 16 + hi * 8;
            for (int i = 0; i < 8; ++i) {
                int r = rbase + i;
                if (r >= M) break;
                float v = acc[a][b][i] * sc + bi;
                if (relu) v = v > 0.f ? v : 0.f;
                if (resid) v += resid[(size_t)r * N + col];
                if (outF) outF[(size_t)r * N + col] = v;
                if (outH) outH[(size_t)r * N + col] = (_Float16)v;
            }
        }
    }
}

// --------------------------------------------------------------- weight prep
__global__ void k_prep_wt(const float* __restrict__ W, _Float16* __restrict__ WT,
                          int Kin, int Nin, int Kpad)
{
    int id = blockIdx.x * 256 + threadIdx.x;
    if (id >= Nin * Kpad) return;
    int n = id / Kpad, k = id % Kpad;
    WT[(size_t)n * Kpad + k] = (k < Kin) ? (_Float16)W[(size_t)k * Nin + n] : (_Float16)0.f;
}

__global__ void k_fold_bn(const float* __restrict__ b, const float* __restrict__ gamma,
                          const float* __restrict__ beta, const float* __restrict__ mean,
                          const float* __restrict__ var, float* sc, float* bi, int n)
{
    int i = blockIdx.x * 256 + threadIdx.x;
    if (i >= n) return;
    float s = gamma[i] * rsqrtf(var[i] + 1e-5f);
    sc[i] = s;
    bi[i] = (b[i] - mean[i]) * s + beta[i];
}

// --------------------------------------------- gather(idx0) + fc1a/relu/fc1b
__global__ __launch_bounds__(128)
void k_gather_mlp0(const float* __restrict__ x, const int* __restrict__ idx0,
                   const float* __restrict__ W1, const float* __restrict__ b1,
                   const float* __restrict__ W2, const float* __restrict__ b2,
                   float* __restrict__ xyz, float* __restrict__ fF,
                   _Float16* __restrict__ fH, int Npts, int total)
{
    __shared__ float sW1[96], sb1[32], sW2[1024], sb2[32];
    for (int i = threadIdx.x; i < 96; i += 128) sW1[i] = W1[i];
    for (int i = threadIdx.x; i < 32; i += 128) { sb1[i] = b1[i]; sb2[i] = b2[i]; }
    for (int i = threadIdx.x; i < 1024; i += 128) sW2[i] = W2[i];
    __syncthreads();
    int pm = blockIdx.x * 128 + threadIdx.x;
    if (pm >= total) return;
    int b = pm / Npts;
    int src = b * Npts + idx0[pm];
    float px = x[src * 3], py = x[src * 3 + 1], pz = x[src * 3 + 2];
    xyz[pm * 3] = px; xyz[pm * 3 + 1] = py; xyz[pm * 3 + 2] = pz;
    float tr[32];
    #pragma unroll
    for (int c = 0; c < 32; ++c) {
        float v = px * sW1[c] + py * sW1[32 + c] + pz * sW1[64 + c] + sb1[c];
        tr[c] = v > 0.f ? v : 0.f;
    }
    #pragma unroll
    for (int c = 0; c < 32; ++c) {
        float h = sb2[c];
        #pragma unroll
        for (int k = 0; k < 32; ++k) h += tr[k] * sW2[k * 32 + c];
        fF[(size_t)pm * 32 + c] = h;
        fH[(size_t)pm * 32 + c] = (_Float16)h;
    }
}

// ----------------------------------------------------------------------- KNN
__global__ __launch_bounds__(128)
void k_knn(const float* __restrict__ q, const float* __restrict__ s,
           int Mq, int Ns, int* __restrict__ nidx)
{
    int b = blockIdx.y;
    int qi = blockIdx.x * 128 + threadIdx.x;
    bool valid = qi < Mq;
    float qx = 0.f, qy = 0.f, qz = 0.f;
    if (valid) {
        const float* p = q + ((size_t)b * Mq + qi) * 3;
        qx = p[0]; qy = p[1]; qz = p[2];
    }
    float bd[16]; int bix[16];
    #pragma unroll
    for (int u = 0; u < 16; ++u) { bd[u] = 3.4e38f; bix[u] = 0; }
    __shared__ float sx[128 * 3];
    for (int t0 = 0; t0 < Ns; t0 += 128) {
        int cnt = (Ns - t0) < 128 ? (Ns - t0) : 128;
        __syncthreads();
        for (int i = threadIdx.x; i < cnt * 3; i += 128)
            sx[i] = s[((size_t)b * Ns + t0) * 3 + i];
        __syncthreads();
        if (valid) {
            for (int j = 0; j < cnt; ++j) {
                float dx = qx - sx[j * 3], dy = qy - sx[j * 3 + 1], dz = qz - sx[j * 3 + 2];
                float d = dx * dx + dy * dy + dz * dz;
                if (d < bd[15]) {
                    float dd = d; int ii = t0 + j;
                    #pragma unroll
                    for (int u = 0; u < 16; ++u) {
                        if (dd < bd[u]) {
                            float td = bd[u]; int ti = bix[u];
                            bd[u] = dd; bix[u] = ii; dd = td; ii = ti;
                        }
                    }
                }
            }
        }
    }
    if (valid) {
        int* o = nidx + ((size_t)b * Mq + qi) * 16;
        #pragma unroll
        for (int u = 0; u < 16; ++u) o[u] = bix[u];
    }
}

// -------------------------------------------- pos_h = relu((xyz_m - xyz_n)@d1)
__global__ void k_d1(const float* __restrict__ xyz, const int* __restrict__ nidx,
                     const float* __restrict__ W, const float* __restrict__ bia,
                     _Float16* __restrict__ posh, int M, int baseP, int rows)
{
    int id = blockIdx.x * 256 + threadIdx.x;
    if (id >= rows * 256) return;
    int r = id >> 8, f = id & 255;
    int gk = baseP * 16 + r;
    int pm = gk >> 4, j = gk & 15;
    int b = pm / M;
    int src = b * M + nidx[(size_t)pm * 16 + j];
    float rx = xyz[pm * 3] - xyz[src * 3];
    float ry = xyz[pm * 3 + 1] - xyz[src * 3 + 1];
    float rz = xyz[pm * 3 + 2] - xyz[src * 3 + 2];
    float v = rx * W[f] + ry * W[256 + f] + rz * W[512 + f] + bia[f];
    posh[(size_t)r * 256 + f] = (_Float16)(v > 0.f ? v : 0.f);
}

// ------------------------------------------------- a_in = q[m] - k[n] + pos
__global__ void k_ain(const _Float16* __restrict__ qH, const _Float16* __restrict__ xkH,
                      const _Float16* __restrict__ pos, const int* __restrict__ nidx,
                      int M, int baseP, int rows, _Float16* __restrict__ ain)
{
    int id = blockIdx.x * 256 + threadIdx.x;
    if (id >= rows * 256) return;
    int r = id >> 8, f = id & 255;
    int gk = baseP * 16 + r;
    int pm = gk >> 4, j = gk & 15;
    int b = pm / M;
    int src = b * M + nidx[(size_t)pm * 16 + j];
    float v = (float)qH[(size_t)pm * 256 + f] - (float)xkH[(size_t)src * 256 + f]
            + (float)pos[(size_t)r * 256 + f];
    ain[(size_t)r * 256 + f] = (_Float16)v;
}

// ------------------- softmax over K=16 neighbors then res = sum a*(v+pos)
__global__ void k_smax(const float* __restrict__ attn, const _Float16* __restrict__ xv,
                       const _Float16* __restrict__ pos, const int* __restrict__ nidx,
                       int M, int baseP, int pts,
                       float* __restrict__ resF, _Float16* __restrict__ resH)
{
    int id = blockIdx.x * 256 + threadIdx.x;
    if (id >= pts * 256) return;
    int pl = id >> 8, f = id & 255;
    int pm = baseP + pl;
    int b = pm / M;
    float lg[16];
    float mx = -3.4e38f;
    #pragma unroll
    for (int j = 0; j < 16; ++j) {
        lg[j] = attn[(size_t)(pl * 16 + j) * 256 + f] * 0.0625f;  // 1/sqrt(256)
        mx = fmaxf(mx, lg[j]);
    }
    float s = 0.f;
    #pragma unroll
    for (int j = 0; j < 16; ++j) { lg[j] = __expf(lg[j] - mx); s += lg[j]; }
    float inv = 1.f / s, acc = 0.f;
    #pragma unroll
    for (int j = 0; j < 16; ++j) {
        int src = b * M + nidx[(size_t)pm * 16 + j];
        float vv = (float)xv[(size_t)src * 256 + f] + (float)pos[(size_t)(pl * 16 + j) * 256 + f];
        acc += lg[j] * inv * vv;
    }
    resF[(size_t)pm * 256 + f] = acc;
    resH[(size_t)pm * 256 + f] = (_Float16)acc;
}

// ------------------------------------------------------ transition-down bits
__global__ void k_gather3(const float* __restrict__ ox, const int* __restrict__ idx,
                          float* __restrict__ nx, int Mn, int Ns, int total)
{
    int pm = blockIdx.x * 256 + threadIdx.x;
    if (pm >= total) return;
    int b = pm / Mn;
    int src = b * Ns + idx[pm];
    nx[pm * 3] = ox[src * 3]; nx[pm * 3 + 1] = ox[src * 3 + 1]; nx[pm * 3 + 2] = ox[src * 3 + 2];
}

__global__ void k_build_h(const float* __restrict__ oxyz, const float* __restrict__ nxyz,
                          const float* __restrict__ featsF, const int* __restrict__ nidx,
                          int Mn, int Ns, int dPrev, int cinPad, int rows,
                          _Float16* __restrict__ h)
{
    int id = blockIdx.x * 256 + threadIdx.x;
    if (id >= rows * cinPad) return;
    int r = id / cinPad, c = id % cinPad;
    int pm = r >> 4, j = r & 15;
    int b = pm / Mn;
    int src = b * Ns + nidx[(size_t)pm * 16 + j];
    float v = 0.f;
    if (c < 3) v = oxyz[src * 3 + c] - nxyz[pm * 3 + c];
    else if (c < 3 + dPrev) v = featsF[(size_t)src * dPrev + (c - 3)];
    h[(size_t)r * cinPad + c] = (_Float16)v;
}

__global__ void k_maxpool(const float* __restrict__ h2, float* __restrict__ fF,
                          _Float16* __restrict__ fH, int BMn, int ch)
{
    int id = blockIdx.x * 256 + threadIdx.x;
    if (id >= BMn * ch) return;
    int pm = id / ch, c = id % ch;
    float m = -3.4e38f;
    #pragma unroll
    for (int j = 0; j < 16; ++j)
        m = fmaxf(m, h2[(size_t)(pm * 16 + j) * ch + c]);
    fF[(size_t)pm * ch + c] = m;
    fH[(size_t)pm * ch + c] = (_Float16)m;
}

// ------------------------------------------------------------------- driver
static inline void gemm(hipStream_t st, const _Float16* A, const _Float16* WT,
                        int M, int N, int Kp, const float* scale, const float* bias,
                        const float* resid, int relu, float* oF, _Float16* oH)
{
    dim3 g((M + TM - 1) / TM, (N + TN - 1) / TN);
    k_gemm<<<g, 256, 0, st>>>(A, WT, M, N, Kp, scale, bias, resid, relu, oF, oH);
}

extern "C" void kernel_launch(void* const* d_in, const int* in_sizes, int n_in,
                              void* d_out, int out_size, void* d_ws, size_t ws_size,
                              hipStream_t stream)
{
    (void)in_sizes; (void)n_in; (void)out_size; (void)ws_size;
    const int B = 4, N = 4096, KNN = 16;

    // d_in layout assumption: setup_inputs() insertion order, params flattened
    // recursively in make_params()/make_transformer() insertion order:
    //  x, idx0..idx3, fc1a{W,b}, fc1b{W,b}, t1{fc1,fc2,d1,d2,g1,g2 (W,b each), wq,wk,wv},
    //  tds[0..2]{conv0{W,b,gamma,beta,mean,var}, conv1{...}}, tfs[0..2]{same as t1}.
    const float* x = (const float*)d_in[0];
    const int* idxs[4] = {(const int*)d_in[1], (const int*)d_in[2],
                          (const int*)d_in[3], (const int*)d_in[4]};
    int pi = 5;
    auto F = [&]() -> const float* { return (const float*)d_in[pi++]; };
    struct Lin { const float* W; const float* b; };
    struct TFp { Lin fc1, fc2, d1, d2, g1, g2; const float *wq, *wk, *wv; };
    struct Conv { const float *W, *b, *gamma, *beta, *mean, *var; };
    Lin fc1a{F(), F()}, fc1b{F(), F()};
    auto readTF = [&]() {
        TFp t;
        t.fc1 = {F(), F()}; t.fc2 = {F(), F()}; t.d1 = {F(), F()};
        t.d2 = {F(), F()};  t.g1 = {F(), F()};  t.g2 = {F(), F()};
        t.wq = F(); t.wk = F(); t.wv = F();
        return t;
    };
    TFp t1 = readTF();
    Conv convs[3][2];
    for (int i = 0; i < 3; ++i)
        for (int j = 0; j < 2; ++j)
            convs[i][j] = {F(), F(), F(), F(), F(), F()};
    TFp tfs[3];
    for (int i = 0; i < 3; ++i) tfs[i] = readTF();

    // Workspace bump allocator.
    char* ws = (char*)d_ws;
    size_t off = 0;
    auto alloc = [&](size_t bytes) -> char* {
        char* p = ws + off;
        off = (off + bytes + 255) & ~(size_t)255;
        return p;
    };
    const int BM0 = B * N;
    float*    xyzA   = (float*)alloc((size_t)BM0 * 3 * 4);
    float*    xyzB   = (float*)alloc((size_t)BM0 * 3 * 4);
    float*    featsF = (float*)alloc((size_t)BM0 * 256 * 4);
    _Float16* featsH = (_Float16*)alloc((size_t)BM0 * 256 * 2);
    _Float16* xH     = (_Float16*)alloc((size_t)BM0 * 256 * 2);
    _Float16* qH     = (_Float16*)alloc((size_t)BM0 * 256 * 2);
    _Float16* xkH    = (_Float16*)alloc((size_t)BM0 * 256 * 2);
    _Float16* xvH    = (_Float16*)alloc((size_t)BM0 * 256 * 2);
    float*    resF   = (float*)alloc((size_t)BM0 * 256 * 4);
    _Float16* resH   = (_Float16*)alloc((size_t)BM0 * 256 * 2);
    int*      nidx   = (int*)alloc((size_t)BM0 * KNN * 4);
    const int CP = 1024, CR = CP * KNN;                 // 16384 K-rows / chunk
    _Float16* poshC = (_Float16*)alloc((size_t)CR * 256 * 2);
    _Float16* posC  = (_Float16*)alloc((size_t)CR * 256 * 2);
    _Float16* ainC  = (_Float16*)alloc((size_t)CR * 256 * 2);
    _Float16* ahC   = (_Float16*)alloc((size_t)CR * 256 * 2);
    float*    attnC = (float*)alloc((size_t)CR * 256 * 4);
    // TD buffers alias the chunk arena (temporally disjoint).
    _Float16* hTD  = poshC;   // max 65536*64 halfs  = 8MB  OK
    _Float16* h1TD = posC;    // max 65536*64 halfs  = 8MB  OK
    float*    h2TD = attnC;   // max 65536*64 floats = 16MB OK

    // Weight prep (f16 transpose + pad, BN fold).
    auto prepLin = [&](const float* W, int Kin, int Nin) -> _Float16* {
        int Kp = (Kin + 31) & ~31;
        _Float16* wt = (_Float16*)alloc((size_t)Nin * Kp * 2);
        int tot = Nin * Kp;
        k_prep_wt<<<(tot + 255) / 256, 256, 0, stream>>>(W, wt, Kin, Nin, Kp);
        return wt;
    };
    struct TFw { _Float16 *fc1, *fc2, *d2, *g1, *g2, *wq, *wk, *wv; };
    auto prepTF = [&](const TFp& t, int d) {
        TFw w;
        w.fc1 = prepLin(t.fc1.W, d, 256);
        w.fc2 = prepLin(t.fc2.W, 256, d);
        w.d2  = prepLin(t.d2.W, 256, 256);
        w.g1  = prepLin(t.g1.W, 256, 256);
        w.g2  = prepLin(t.g2.W, 256, 256);
        w.wq  = prepLin(t.wq, 256, 256);
        w.wk  = prepLin(t.wk, 256, 256);
        w.wv  = prepLin(t.wv, 256, 256);
        return w;
    };
    TFw w1 = prepTF(t1, 32);
    TFw wtfs[3];
    for (int i = 0; i < 3; ++i) wtfs[i] = prepTF(tfs[i], 64 << i);
    _Float16* wConv[3][2]; float* convSc[3][2]; float* convBi[3][2];
    for (int i = 0; i < 3; ++i) {
        int dPrev = 32 << i, ch = 64 << i, cin = 3 + dPrev;
        wConv[i][0] = prepLin(convs[i][0].W, cin, ch);
        wConv[i][1] = prepLin(convs[i][1].W, ch, ch);
        for (int j = 0; j < 2; ++j) {
            convSc[i][j] = (float*)alloc((size_t)ch * 4);
            convBi[i][j] = (float*)alloc((size_t)ch * 4);
            k_fold_bn<<<(ch + 255) / 256, 256, 0, stream>>>(
                convs[i][j].b, convs[i][j].gamma, convs[i][j].beta,
                convs[i][j].mean, convs[i][j].var, convSc[i][j], convBi[i][j], ch);
        }
    }

    // Stage 0: gather by idx0 + tiny MLP (3->32->32).
    k_gather_mlp0<<<(BM0 + 127) / 128, 128, 0, stream>>>(
        x, idxs[0], fc1a.W, fc1a.b, fc1b.W, fc1b.b, xyzA, featsF, featsH, N, BM0);

    float* xyzCur = xyzA;
    float* xyzNxt = xyzB;
    int M = N, d = 32;

    auto runTF = [&](const TFp& t, const TFw& w, bool last) {
        int BM = B * M;
        k_knn<<<dim3((M + 127) / 128, B), 128, 0, stream>>>(xyzCur, xyzCur, M, M, nidx);
        gemm(stream, featsH, w.fc1, BM, 256, d,   nullptr, t.fc1.b, nullptr, 0, nullptr, xH);
        gemm(stream, xH, w.wq, BM, 256, 256, nullptr, nullptr, nullptr, 0, nullptr, qH);
        gemm(stream, xH, w.wk, BM, 256, 256, nullptr, nullptr, nullptr, 0, nullptr, xkH);
        gemm(stream, xH, w.wv, BM, 256, 256, nullptr, nullptr, nullptr, 0, nullptr, xvH);
        for (int base = 0; base < BM; base += CP) {
            int pts = (BM - base) < CP ? (BM - base) : CP;
            int rows = pts * KNN;
            int tot = rows * 256;
            k_d1<<<(tot + 255) / 256, 256, 0, stream>>>(
                xyzCur, nidx, t.d1.W, t.d1.b, poshC, M, base, rows);
            gemm(stream, poshC, w.d2, rows, 256, 256, nullptr, t.d2.b, nullptr, 0, nullptr, posC);
            k_ain<<<(tot + 255) / 256, 256, 0, stream>>>(
                qH, xkH, posC, nidx, M, base, rows, ainC);
            gemm(stream, ainC, w.g1, rows, 256, 256, nullptr, t.g1.b, nullptr, 1, nullptr, ahC);
            gemm(stream, ahC,  w.g2, rows, 256, 256, nullptr, t.g2.b, nullptr, 0, attnC, nullptr);
            int tot2 = pts * 256;
            k_smax<<<(tot2 + 255) / 256, 256, 0, stream>>>(
                attnC, xvH, posC, nidx, M, base, pts, resF, resH);
        }
        gemm(stream, resH, w.fc2, BM, d, 256, nullptr, t.fc2.b, featsF, 0,
             last ? (float*)d_out : featsF, featsH);
    };

    runTF(t1, w1, false);
    for (int i = 0; i < 3; ++i) {
        int Ns = M, Mn = M / 4, dPrev = d, ch = 2 * d;
        int cin = 3 + dPrev, cinPad = (cin + 31) & ~31;
        int BMn = B * Mn, rows = BMn * KNN;
        k_gather3<<<(BMn + 255) / 256, 256, 0, stream>>>(xyzCur, idxs[i + 1], xyzNxt, Mn, Ns, BMn);
        k_knn<<<dim3((Mn + 127) / 128, B), 128, 0, stream>>>(xyzNxt, xyzCur, Mn, Ns, nidx);
        int tot = rows * cinPad;
        k_build_h<<<(tot + 255) / 256, 256, 0, stream>>>(
            xyzCur, xyzNxt, featsF, nidx, Mn, Ns, dPrev, cinPad, rows, hTD);
        gemm(stream, hTD,  wConv[i][0], rows, ch, cinPad, convSc[i][0], convBi[i][0],
             nullptr, 1, nullptr, h1TD);
        gemm(stream, h1TD, wConv[i][1], rows, ch, ch, convSc[i][1], convBi[i][1],
             nullptr, 1, h2TD, nullptr);
        int tot2 = BMn * ch;
        k_maxpool<<<(tot2 + 255) / 256, 256, 0, stream>>>(h2TD, featsF, featsH, BMn, ch);
        float* tmp = xyzCur; xyzCur = xyzNxt; xyzNxt = tmp;
        M = Mn; d = ch;
        runTF(tfs[i], wtfs[i], i == 2);
    }
}
